// RopeAttention_36764920054216
// MI455X (gfx1250) — matrix operand order repaired and checked by reading, not compiled, not run on hardware
//
#include <hip/hip_runtime.h>
#include <hip/hip_bf16.h>
#include <math.h>

// ---------------- problem constants ----------------
#define BB   4
#define TT   2048
#define CC   1024
#define HH   16
#define HD   64
#define MM   (BB * TT)        // 8192 rows
#define N1   (3 * CC)         // 3072 qkv cols

typedef __attribute__((ext_vector_type(16))) _Float16 v16h;
typedef __attribute__((ext_vector_type(8)))  float    v8f;
typedef unsigned int u32x4 __attribute__((ext_vector_type(4)));
typedef int          i32x8 __attribute__((ext_vector_type(8)));
typedef int          i32x4 __attribute__((ext_vector_type(4)));

// ---------------- WMMA helpers (CDNA5, wave32) ----------------
__device__ __forceinline__ v8f wmma16x16x32(v16h a, v16h b, v8f c) {
    return __builtin_amdgcn_wmma_f32_16x16x32_f16(
        false, a, false, b, (short)0, c, false, false);
}

__device__ __forceinline__ v8f zero8() {
    v8f z = {0.f, 0.f, 0.f, 0.f, 0.f, 0.f, 0.f, 0.f};
    return z;
}

// A-matrix 16x32 f16 fragment from row-major [16][>=32] storage (stride ld halfs).
// ISA: lane = row (mod 16); VGPR v holds K = (v&3)*2 + (v>>2)*16 + (lane>>4)*8.
__device__ __forceinline__ v16h load_a_frag(const _Float16* base, int ld, int lane) {
    v16h a;
    const int row  = lane & 15;
    const int koff = (lane >> 4) << 3;
    const _Float16* p = base + row * ld + koff;
#pragma unroll
    for (int v = 0; v < 8; ++v) {
        int k = ((v & 3) << 1) + ((v >> 2) << 4);
        a[2 * v]     = p[k];
        a[2 * v + 1] = p[k + 1];
    }
    return a;
}

// B-matrix 32x16 f16 fragment from K-contiguous storage: element (k, col) at
// base[col*ld + k]. ISA: lane = col (mod 16); VGPR v holds K = (v&3)*2 + (v>>2)*16 + (lane>>4)*8.
__device__ __forceinline__ v16h load_b_frag(const _Float16* base, int ld, int lane) {
    v16h b;
    const int col  = lane & 15;
    const int koff = (lane >> 4) << 3;
    const _Float16* p = base + col * ld + koff;
#pragma unroll
    for (int v = 0; v < 8; ++v) {
        int k = ((v & 3) << 1) + ((v >> 2) << 4);
        b[2 * v]     = p[k];
        b[2 * v + 1] = p[k + 1];
    }
    return b;
}

// ---------------- Tensor Data Mover: 2-D f16 tile -> LDS ----------------
// D# per CDNA5 ISA. Group0: count=1, lds_addr, global
// addr, type=2. Group1: data_size=1 (2 bytes), tensor_dim0=row stride,
// tensor_dim1=tile_h, tile_dim0/1, tensor_dim0_stride. Groups 2/3 zero (2-D).
__device__ __forceinline__ void tdm_load_2d_f16(
    unsigned int lds_addr, const _Float16* gptr,
    unsigned int tile_w, unsigned int tile_h, unsigned int stride_elems)
{
    unsigned long long ga = (unsigned long long)(uintptr_t)gptr;
    u32x4 g0;
    g0[0] = 1u;                                        // count=1 (valid), user mode
    g0[1] = lds_addr;                                  // LDS byte address
    g0[2] = (unsigned int)ga;                          // global_addr[31:0]
    g0[3] = (unsigned int)((ga >> 32) & 0x1FFFFFFu)    // global_addr[56:32]
          | (2u << 30);                                // type=2 ("image")
    i32x8 g1;
    g1[0] = (int)(1u << 16);                           // data_size=1 -> 2 bytes
    g1[1] = (int)((stride_elems & 0xFFFFu) << 16);     // tensor_dim0[15:0]
    g1[2] = (int)((stride_elems >> 16)                 // tensor_dim0[31:16]
          | ((tile_h & 0xFFFFu) << 16));               // tensor_dim1[15:0]
    g1[3] = (int)((tile_h >> 16)                       // tensor_dim1[31:16]
          | (tile_w << 16));                           // tile_dim0
    g1[4] = (int)(tile_h & 0xFFFFu);                   // tile_dim1 (tile_dim2=0)
    g1[5] = (int)stride_elems;                         // tensor_dim0_stride[31:0]
    g1[6] = 0;                                         // stride[47:32], dim1_stride lo
    g1[7] = 0;
    i32x4 z4 = {0, 0, 0, 0};
#if __clang_major__ >= 23
    i32x8 z8 = {0, 0, 0, 0, 0, 0, 0, 0};
    __builtin_amdgcn_tensor_load_to_lds(g0, g1, z4, z4, z8, 0);
#else
    __builtin_amdgcn_tensor_load_to_lds(g0, g1, z4, z4, 0);
#endif
}

__device__ __forceinline__ unsigned int lds_addr_of(const void* p) {
    return (unsigned int)(uintptr_t)p;   // LDS aperture: low 32 bits = LDS offset
}

// ---------------- prep kernels ----------------
__global__ void k_convert_f32_to_f16(const float* __restrict__ src,
                                     _Float16* __restrict__ dst, int n4) {
    int i = blockIdx.x * blockDim.x + threadIdx.x;
    if (i < n4) {
        float4 v = ((const float4*)src)[i];
        dst[4 * i + 0] = (_Float16)v.x;
        dst[4 * i + 1] = (_Float16)v.y;
        dst[4 * i + 2] = (_Float16)v.z;
        dst[4 * i + 3] = (_Float16)v.w;
    }
}

// Wt[n][k] = (f16) W[k][n]
__global__ void k_transpose_to_f16(const float* __restrict__ W,
                                   _Float16* __restrict__ Wt, int K, int N) {
    __shared__ float tile[32][33];
    const int tx = threadIdx.x;
    const int ty = threadIdx.y;
    const int bn = blockIdx.x * 32;
    const int bk = blockIdx.y * 32;
#pragma unroll
    for (int j = 0; j < 4; ++j) {
        int k = bk + ty + j * 8;
        tile[ty + j * 8][tx] = W[(size_t)k * N + bn + tx];
    }
    __syncthreads();
#pragma unroll
    for (int j = 0; j < 4; ++j) {
        int n = bn + ty + j * 8;
        Wt[(size_t)n * K + bk + tx] = (_Float16)tile[tx][ty + j * 8];
    }
}

__global__ void k_rope_tables(float* __restrict__ cosT, float* __restrict__ sinT) {
    const int t = blockIdx.x;
    const int d = threadIdx.x;
    float expo = (float)(d & ~1) * (1.0f / (float)HD);
    float invf = powf(10000.0f, -expo);
    float ang  = (float)t * invf;
    cosT[t * HD + d] = cosf(ang);
    sinT[t * HD + d] = sinf(ang);
}

// ---------------- fused QKV GEMM + bias + RoPE + head scatter ----------------
__global__ __launch_bounds__(256) void k_qkv_gemm_rope(
    const _Float16* __restrict__ Xh,     // [M][C]
    const _Float16* __restrict__ Wt,     // [N1][C]
    const float* __restrict__ bqkv,
    const float* __restrict__ cosT, const float* __restrict__ sinT,
    _Float16* __restrict__ Qh, _Float16* __restrict__ Kh, _Float16* __restrict__ Vh)
{
    __shared__ __align__(16) _Float16 As[2][128 * 32];
    __shared__ __align__(16) _Float16 Bs[2][256 * 32];

    const int tid  = threadIdx.x;
    const int lane = tid & 31;
    const int wid  = tid >> 5;
    const int wm   = wid >> 2;
    const int wn   = wid & 3;
    const int blockM = blockIdx.y * 128;
    const int blockN = blockIdx.x * 256;

    const _Float16* Ag = Xh + (size_t)blockM * CC;
    const _Float16* Bg = Wt + (size_t)blockN * CC;

    v8f acc[4][4];
#pragma unroll
    for (int m = 0; m < 4; ++m)
#pragma unroll
        for (int n = 0; n < 4; ++n) acc[m][n] = zero8();

    if (wid == 0) {   // TDM prefetch of tile 0 (wave-level DMA, EXEC ignored)
        tdm_load_2d_f16(lds_addr_of(&As[0][0]), Ag, 32, 128, CC);
        tdm_load_2d_f16(lds_addr_of(&Bs[0][0]), Bg, 32, 256, CC);
    }

    for (int i = 0; i < CC / 32; ++i) {
        const int buf = i & 1;
        if (wid == 0) __builtin_amdgcn_s_wait_tensorcnt(0);  // tile i landed
        __syncthreads();   // all waves: tile i visible, tile i-1 reads done
        if (wid == 0 && i + 1 < CC / 32) {                   // DMA tile i+1
            tdm_load_2d_f16(lds_addr_of(&As[buf ^ 1][0]), Ag + (i + 1) * 32, 32, 128, CC);
            tdm_load_2d_f16(lds_addr_of(&Bs[buf ^ 1][0]), Bg + (i + 1) * 32, 32, 256, CC);
        }
        v16h af[4], bf[4];
#pragma unroll
        for (int n = 0; n < 4; ++n) bf[n] = load_b_frag(&Bs[buf][(wn * 64 + n * 16) * 32], 32, lane);
#pragma unroll
        for (int m = 0; m < 4; ++m) af[m] = load_a_frag(&As[buf][(wm * 64 + m * 16) * 32], 32, lane);
#pragma unroll
        for (int m = 0; m < 4; ++m)
#pragma unroll
            for (int n = 0; n < 4; ++n) acc[m][n] = wmma16x16x32(af[m], bf[n], acc[m][n]);
    }

    // epilogue: bias + RoPE(q,k) + scatter to head-major f16
    const int row0 = blockM + wm * 64;
    const int col0 = blockN + wn * 64;
#pragma unroll
    for (int m = 0; m < 4; ++m) {
#pragma unroll
        for (int n = 0; n < 4; ++n) {
            const int gc   = col0 + n * 16 + (lane & 15);
            const float bias = bqkv[gc];
            const int sec = gc >> 10;          // 0=q, 1=k, 2=v (uniform per subtile)
            const int win = gc & 1023;
            const int h   = win >> 6;
            const int d   = win & 63;
            _Float16* dst = (sec == 0) ? Qh : ((sec == 1) ? Kh : Vh);
            const float qs = (sec == 0) ? 0.125f : 1.0f;   // 1/sqrt(64) into q
#pragma unroll
            for (int r = 0; r < 8; ++r) {
                const int grow = row0 + m * 16 + r + ((lane >> 4) << 3);
                const int b = grow >> 11;
                const int t = grow & (TT - 1);
                float val = acc[m][n][r] + bias;
                float outv;
                if (sec < 2) {
                    float partner = __shfl_xor(val, 1, 32);
                    float rh = (d & 1) ? partner : -partner;   // rotate_half
                    outv = (val * cosT[t * HD + d] + rh * sinT[t * HD + d]) * qs;
                } else {
                    outv = val;
                }
                dst[(((size_t)(b * HH + h) * TT + t) << 6) + d] = (_Float16)outv;
            }
        }
    }
}

// ---------------- flash attention (full-key softmax + additive mask) ----------------
__global__ __launch_bounds__(256) void k_attention(
    const _Float16* __restrict__ Qh, const _Float16* __restrict__ Kh,
    const _Float16* __restrict__ Vh, const float* __restrict__ pad,
    _Float16* __restrict__ Yh)
{
    __shared__ __align__(16) _Float16 Ks[64 * 64];       // [key][d]  (TDM)
    __shared__ __align__(16) _Float16 Vt[64 * 64];       // [d][key]  (manual transpose)
    __shared__ __align__(16) _Float16 Ps[8 * 16 * 64];   // per-wave P tile

    const int tid  = threadIdx.x;
    const int lane = tid & 31;
    const int wid  = tid >> 5;
    const int bh   = blockIdx.y;
    const int b    = bh >> 4;
    const int h    = bh & 15;
    const int q0   = blockIdx.x * 128 + wid * 16;
    const size_t headBase = (size_t)bh * TT * HD;

    v16h aq0 = load_a_frag(Qh + headBase + (size_t)q0 * HD,      HD, lane);
    v16h aq1 = load_a_frag(Qh + headBase + (size_t)q0 * HD + 32, HD, lane);

    v8f accO[4];
#pragma unroll
    for (int n = 0; n < 4; ++n) accO[n] = zero8();
    float m_i[8], l_i[8];
#pragma unroll
    for (int r = 0; r < 8; ++r) { m_i[r] = -1e30f; l_i[r] = 0.f; }

    _Float16* Pw = &Ps[wid * 16 * 64];

    for (int kt = 0; kt < TT; kt += 64) {
        if (wid == 0)   // K tile via Tensor Data Mover (64x64, contiguous rows)
            tdm_load_2d_f16(lds_addr_of(&Ks[0]), Kh + headBase + (size_t)kt * HD, 64, 64, 64);
        {   // V tile transposed (manual; TDM cannot transpose) -- overlaps TDM
            int key = tid >> 2;
            int d0  = (tid & 3) * 16;
            const _Float16* src = Vh + headBase + (size_t)(kt + key) * HD + d0;
            if (kt + 64 < TT)
                __builtin_prefetch(src + 64 * HD, 0, 0);   // global_prefetch_b8
            v16h tmp = *(const v16h*)src;
#pragma unroll
            for (int j = 0; j < 16; ++j) Vt[(d0 + j) * 64 + key] = tmp[j];
        }
        if (wid == 0) __builtin_amdgcn_s_wait_tensorcnt(0);
        __syncthreads();

        // S = Q * K^T
        v8f sacc[4];
#pragma unroll
        for (int n = 0; n < 4; ++n) {
            v16h bk0 = load_b_frag(&Ks[(n * 16) * 64 + 0],  64, lane);
            v16h bk1 = load_b_frag(&Ks[(n * 16) * 64 + 32], 64, lane);
            sacc[n] = wmma16x16x32(aq0, bk0, zero8());
            sacc[n] = wmma16x16x32(aq1, bk1, sacc[n]);
        }

        // additive mask: + (k<=q ? pad[b][k] : 0)
#pragma unroll
        for (int n = 0; n < 4; ++n) {
            const int kg = kt + n * 16 + (lane & 15);
            const float padv = pad[b * TT + kg];
#pragma unroll
            for (int r = 0; r < 8; ++r) {
                const int qg = q0 + r + ((lane >> 4) << 3);
                sacc[n][r] += (kg <= qg) ? padv : 0.f;
            }
        }

        // online softmax, per-row stats via 16-lane shuffle reductions
#pragma unroll
        for (int r = 0; r < 8; ++r) {
            float v = fmaxf(fmaxf(sacc[0][r], sacc[1][r]), fmaxf(sacc[2][r], sacc[3][r]));
#pragma unroll
            for (int off = 1; off < 16; off <<= 1) v = fmaxf(v, __shfl_xor(v, off, 32));
            float mn = fmaxf(m_i[r], v);
            float sc = __expf(m_i[r] - mn);
            m_i[r] = mn;
            l_i[r] *= sc;
#pragma unroll
            for (int n = 0; n < 4; ++n) accO[n][r] *= sc;
        }

        float psum[8];
#pragma unroll
        for (int r = 0; r < 8; ++r) psum[r] = 0.f;
#pragma unroll
        for (int n = 0; n < 4; ++n) {
#pragma unroll
            for (int r = 0; r < 8; ++r) {
                float p = __expf(sacc[n][r] - m_i[r]);
                psum[r] += p;
                Pw[(r + ((lane >> 4) << 3)) * 64 + n * 16 + (lane & 15)] = (_Float16)p;
            }
        }
#pragma unroll
        for (int r = 0; r < 8; ++r) {
            float v = psum[r];
#pragma unroll
            for (int off = 1; off < 16; off <<= 1) v += __shfl_xor(v, off, 32);
            l_i[r] += v;
        }

        // O += P(16x64) * V(64x64)
        v16h ap0 = load_a_frag(&Pw[0],  64, lane);
        v16h ap1 = load_a_frag(&Pw[32], 64, lane);
#pragma unroll
        for (int n = 0; n < 4; ++n) {
            v16h bv0 = load_b_frag(&Vt[(n * 16) * 64 + 0],  64, lane);
            v16h bv1 = load_b_frag(&Vt[(n * 16) * 64 + 32], 64, lane);
            accO[n] = wmma16x16x32(ap0, bv0, accO[n]);
            accO[n] = wmma16x16x32(ap1, bv1, accO[n]);
        }
        __syncthreads();
    }

    float invl[8];
#pragma unroll
    for (int r = 0; r < 8; ++r) invl[r] = 1.0f / l_i[r];
#pragma unroll
    for (int n = 0; n < 4; ++n) {
        const int d = n * 16 + (lane & 15);
#pragma unroll
        for (int r = 0; r < 8; ++r) {
            const int qg = q0 + r + ((lane >> 4) << 3);
            Yh[((size_t)(b * TT + qg) << 10) + h * HD + d] = (_Float16)(accO[n][r] * invl[r]);
        }
    }
}

// ---------------- projection GEMM (f32 output + bias) ----------------
__global__ __launch_bounds__(256) void k_proj_gemm(
    const _Float16* __restrict__ Yh,     // [M][C]
    const _Float16* __restrict__ Wt,     // [C][C]
    const float* __restrict__ bproj,
    float* __restrict__ out)             // [M][C] f32
{
    __shared__ __align__(16) _Float16 As[2][128 * 32];
    __shared__ __align__(16) _Float16 Bs[2][256 * 32];

    const int tid  = threadIdx.x;
    const int lane = tid & 31;
    const int wid  = tid >> 5;
    const int wm   = wid >> 2;
    const int wn   = wid & 3;
    const int blockM = blockIdx.y * 128;
    const int blockN = blockIdx.x * 256;

    const _Float16* Ag = Yh + (size_t)blockM * CC;
    const _Float16* Bg = Wt + (size_t)blockN * CC;

    v8f acc[4][4];
#pragma unroll
    for (int m = 0; m < 4; ++m)
#pragma unroll
        for (int n = 0; n < 4; ++n) acc[m][n] = zero8();

    if (wid == 0) {
        tdm_load_2d_f16(lds_addr_of(&As[0][0]), Ag, 32, 128, CC);
        tdm_load_2d_f16(lds_addr_of(&Bs[0][0]), Bg, 32, 256, CC);
    }

    for (int i = 0; i < CC / 32; ++i) {
        const int buf = i & 1;
        if (wid == 0) __builtin_amdgcn_s_wait_tensorcnt(0);
        __syncthreads();
        if (wid == 0 && i + 1 < CC / 32) {
            tdm_load_2d_f16(lds_addr_of(&As[buf ^ 1][0]), Ag + (i + 1) * 32, 32, 128, CC);
            tdm_load_2d_f16(lds_addr_of(&Bs[buf ^ 1][0]), Bg + (i + 1) * 32, 32, 256, CC);
        }
        v16h af[4], bf[4];
#pragma unroll
        for (int n = 0; n < 4; ++n) bf[n] = load_b_frag(&Bs[buf][(wn * 64 + n * 16) * 32], 32, lane);
#pragma unroll
        for (int m = 0; m < 4; ++m) af[m] = load_a_frag(&As[buf][(wm * 64 + m * 16) * 32], 32, lane);
#pragma unroll
        for (int m = 0; m < 4; ++m)
#pragma unroll
            for (int n = 0; n < 4; ++n) acc[m][n] = wmma16x16x32(af[m], bf[n], acc[m][n]);
    }

    const int row0 = blockM + wm * 64;
    const int col0 = blockN + wn * 64;
#pragma unroll
    for (int m = 0; m < 4; ++m) {
#pragma unroll
        for (int n = 0; n < 4; ++n) {
            const int gc = col0 + n * 16 + (lane & 15);
            const float bias = bproj[gc];
#pragma unroll
            for (int r = 0; r < 8; ++r) {
                const int grow = row0 + m * 16 + r + ((lane >> 4) << 3);
                out[(size_t)grow * CC + gc] = acc[m][n][r] + bias;
            }
        }
    }
}

// ---------------- host launcher ----------------
extern "C" void kernel_launch(void* const* d_in, const int* in_sizes, int n_in,
                              void* d_out, int out_size, void* d_ws, size_t ws_size,
                              hipStream_t stream) {
    (void)in_sizes; (void)n_in; (void)out_size; (void)ws_size;

    const float* x     = (const float*)d_in[0];
    const float* pad   = (const float*)d_in[1];
    const float* Wqkv  = (const float*)d_in[2];
    const float* bqkv  = (const float*)d_in[3];
    const float* Wproj = (const float*)d_in[4];
    const float* bproj = (const float*)d_in[5];
    float* out = (float*)d_out;

    char* w = (char*)d_ws;
    _Float16* Xh     = (_Float16*)w; w += (size_t)MM * CC * 2;
    _Float16* Wqkvt  = (_Float16*)w; w += (size_t)N1 * CC * 2;
    _Float16* Wprojt = (_Float16*)w; w += (size_t)CC * CC * 2;
    _Float16* Qh     = (_Float16*)w; w += (size_t)MM * CC * 2;
    _Float16* Kh     = (_Float16*)w; w += (size_t)MM * CC * 2;
    _Float16* Vh     = (_Float16*)w; w += (size_t)MM * CC * 2;
    _Float16* Yh     = (_Float16*)w; w += (size_t)MM * CC * 2;
    float*    cosT   = (float*)w;    w += (size_t)TT * HD * 4;
    float*    sinT   = (float*)w;    w += (size_t)TT * HD * 4;

    {
        int n4 = (MM * CC) / 4;
        k_convert_f32_to_f16<<<n4 / 256, 256, 0, stream>>>(x, Xh, n4);
    }
    k_transpose_to_f16<<<dim3(N1 / 32, CC / 32), dim3(32, 8), 0, stream>>>(Wqkv, Wqkvt, CC, N1);
    k_transpose_to_f16<<<dim3(CC / 32, CC / 32), dim3(32, 8), 0, stream>>>(Wproj, Wprojt, CC, CC);
    k_rope_tables<<<TT, HD, 0, stream>>>(cosT, sinT);

    k_qkv_gemm_rope<<<dim3(N1 / 256, MM / 128), 256, 0, stream>>>(
        Xh, Wqkvt, bqkv, cosT, sinT, Qh, Kh, Vh);

    k_attention<<<dim3(TT / 128, BB * HH), 256, 0, stream>>>(Qh, Kh, Vh, pad, Yh);

    k_proj_gemm<<<dim3(CC / 256, MM / 128), 256, 0, stream>>>(Yh, Wprojt, bproj, out);
}
